// AlexNetOWT_BN_14929306321792
// MI455X (gfx1250) — 8-bit matrix operand order repaired and checked by reading, compiled, not run on hardware
//
#include <hip/hip_runtime.h>
#include <math.h>

typedef int v2i __attribute__((ext_vector_type(2)));
typedef int v4i __attribute__((ext_vector_type(4)));
typedef int v8i __attribute__((ext_vector_type(8)));

#define BDIM 256

__device__ __forceinline__ float fsgn(float v) {
    return v > 0.f ? 1.f : (v < 0.f ? -1.f : 0.f);
}

__device__ __forceinline__ v8i cat44(v4i a, v4i b) {
    return __builtin_shufflevector(a, b, 0, 1, 2, 3, 4, 5, 6, 7);
}

__global__ void __launch_bounds__(BDIM)
conv1_pool_kernel(const float* __restrict__ x, const float* __restrict__ w,
                  const float* __restrict__ bias, float* __restrict__ out)
{
    long idx = (long)blockIdx.x * blockDim.x + threadIdx.x;
    const long total = 16L * 3196 * 192;
    if (idx >= total) return;
    int o = (int)(idx % 192);
    long r = idx / 192;
    int q = (int)(r % 3196);
    int b = (int)(r / 3196);

    float sw[15];
#pragma unroll
    for (int i = 0; i < 15; i++) sw[i] = fsgn(w[o * 15 + i]);

    const float* xb = x + (long)b * 3 * 15988;
    float m = -3.4e38f;
    for (int i = 0; i < 5; i++) {
        int p = q * 5 - 2 + i;
        if (p < 0 || p >= 15976) continue;
        float s = 0.f;
#pragma unroll
        for (int c = 0; c < 3; c++) {
            const float* xc = xb + (long)c * 15988 + p;
#pragma unroll
            for (int t = 0; t < 5; t++) s += xc[t * 3] * sw[c * 5 + t];
        }
        m = fmaxf(m, s);
    }
    out[idx] = m + bias[o];
}

__global__ void __launch_bounds__(BDIM)
maxpool_kernel(const float* __restrict__ in, float* __restrict__ out,
               int B, int Lin, int Lout, int C, int k, int pad)
{
    long idx = (long)blockIdx.x * blockDim.x + threadIdx.x;
    long total = (long)B * Lout * C;
    if (idx >= total) return;
    int c = (int)(idx % C);
    long r = idx / C;
    int q = (int)(r % Lout);
    int b = (int)(r / Lout);
    float m = -3.4e38f;
    for (int i = 0; i < k; i++) {
        int p = q * k - pad + i;
        if (p < 0 || p >= Lin) continue;
        m = fmaxf(m, in[((long)b * Lin + p) * C + c]);
    }
    out[idx] = m;
}

__global__ void __launch_bounds__(BDIM)
bn_stats_kernel(const float* __restrict__ x, long R, int C,
                float* __restrict__ mean, float* __restrict__ invstd)
{
    int c = blockIdx.x;
    int tid = threadIdx.x;
    float s = 0.f, ss = 0.f;
    for (long r = tid; r < R; r += BDIM) {
        float v = x[r * C + c];
        s += v; ss += v * v;
    }
    __shared__ float sh[BDIM], sh2[BDIM];
    sh[tid] = s; sh2[tid] = ss;
    __syncthreads();
    for (int st = BDIM / 2; st > 0; st >>= 1) {
        if (tid < st) { sh[tid] += sh[tid + st]; sh2[tid] += sh2[tid + st]; }
        __syncthreads();
    }
    if (tid == 0) {
        float m = sh[0] / (float)R;
        float v = sh2[0] / (float)R - m * m;
        mean[c] = m;
        invstd[c] = rsqrtf(v + 1e-5f);
    }
}

__global__ void __launch_bounds__(BDIM)
bn_relu_sign_kernel(const float* __restrict__ x, long total, int C,
                    const float* __restrict__ mean, const float* __restrict__ invstd,
                    const float* __restrict__ g, const float* __restrict__ be,
                    signed char* __restrict__ out)
{
    long idx = (long)blockIdx.x * blockDim.x + threadIdx.x;
    if (idx >= total) return;
    int c = (int)(idx % C);
    float y = (x[idx] - mean[c]) * invstd[c] * g[c] + be[c];
    out[idx] = (y > 0.f) ? (signed char)1 : (signed char)0;
}

__global__ void __launch_bounds__(BDIM)
bn_relu_sign_tr5_kernel(const float* __restrict__ x,
                        const float* __restrict__ mean, const float* __restrict__ invstd,
                        const float* __restrict__ g, const float* __restrict__ be,
                        signed char* __restrict__ out)
{
    long idx = (long)blockIdx.x * blockDim.x + threadIdx.x;
    const long total = 16L * 352 * 72;
    if (idx >= total) return;
    int c = (int)(idx % 72);
    long r = idx / 72;
    int l = (int)(r % 352);
    int b = (int)(r / 352);
    float y = (x[idx] - mean[c]) * invstd[c] * g[c] + be[c];
    signed char s = (y > 0.f) ? (signed char)1 : (signed char)0;
    out[(long)b * 25344 + (long)c * 352 + l] = s;
}

__global__ void __launch_bounds__(BDIM)
pack_w_swz_kernel(const float* __restrict__ w, signed char* __restrict__ dst,
                  int O, int I, int T, int Opad)
{
    int K = I * T;
    int nchunks = K >> 6;
    long total_dw = ((long)Opad * K) >> 2;
    long idx = (long)blockIdx.x * blockDim.x + threadIdx.x;
    if (idx >= total_dw) return;
    int j = (int)(idx & 7);
    int lane = (int)((idx >> 3) & 31);
    long rest = idx >> 8;
    int ch = (int)(rest % nchunks);
    int tile = (int)(rest / nchunks);
    int hi = lane >> 4;
    int lo = lane & 15;
    int row = tile * 16 + lo;
    int k = ch * 64 + ((j >> 2) << 5) + (hi << 4) + ((j & 3) << 2);
    unsigned v = 0u;
    if (row < O) {
        int t = k / I;
        int i = k - t * I;
#pragma unroll
        for (int q = 0; q < 4; q++) {
            float f = w[((long)row * I + (i + q)) * T + t];
            unsigned char sb = (unsigned char)(f > 0.f ? 1 : (f < 0.f ? -1 : 0));
            v |= ((unsigned)sb) << (8 * q);
        }
    }
    ((unsigned*)dst)[idx] = v;
}

__global__ void __launch_bounds__(BDIM)
bin_gemm_wmma(const signed char* __restrict__ Aswz, const signed char* __restrict__ act,
              const float* __restrict__ bias, float* __restrict__ out,
              int M, int Mpad, int Cin, int T, int dil,
              int Lin, int Lout, int Ntotal)
{
    const int lane = threadIdx.x & 31;
    const int lo = lane & 15;
    const int hi = lane >> 4;
    const int wv = __builtin_amdgcn_readfirstlane(threadIdx.x) >> 5;

    const int nchunks = (T * Cin) >> 6;
    int tiles_m2 = Mpad >> 5;
    int tiles_n4 = (Ntotal + 63) >> 6;
    int wtile = blockIdx.x * 8 + wv;
    if (wtile >= tiles_m2 * tiles_n4) return;
    int tm2 = wtile % tiles_m2;
    int n0 = (wtile / tiles_m2) * 64;

    bool on[4];
    long rowb[4];
#pragma unroll
    for (int nt = 0; nt < 4; nt++) {
        on[nt] = (n0 + nt * 16) < Ntotal;
        int col = on[nt] ? (n0 + nt * 16 + lo) : lo;
        int b = col / Lout;
        int p = col - b * Lout;
        rowb[nt] = ((long)b * Lin + p) * (long)Cin + hi * 16;
    }
    const signed char* Ap0 = Aswz + ((long)(tm2 * 2    ) * nchunks) * 1024 + lane * 32;
    const signed char* Ap1 = Aswz + ((long)(tm2 * 2 + 1) * nchunks) * 1024 + lane * 32;

    const v8i z = {0, 0, 0, 0, 0, 0, 0, 0};
    v8i acc[2][4] = {{z, z, z, z}, {z, z, z, z}};

    for (int t = 0; t < T; t++) {
        long tb = (long)t * dil * Cin;
        const signed char* bp0 = act + rowb[0] + tb;
        const signed char* bp1 = act + rowb[1] + tb;
        const signed char* bp2 = act + rowb[2] + tb;
        const signed char* bp3 = act + rowb[3] + tb;
        for (int cc = 0; cc < Cin; cc += 192) {
#pragma unroll
            for (int u = 0; u < 3; u++) {
                const int ao = u * 1024;
                const int bo = u * 64;
                v8i af0 = cat44(*(const v4i*)(Ap0 + ao), *(const v4i*)(Ap0 + ao + 16));
                v8i af1 = cat44(*(const v4i*)(Ap1 + ao), *(const v4i*)(Ap1 + ao + 16));
                v8i bf0 = cat44(*(const v4i*)(bp0 + bo), *(const v4i*)(bp0 + bo + 32));
                v8i bf1 = cat44(*(const v4i*)(bp1 + bo), *(const v4i*)(bp1 + bo + 32));
                v8i bf2 = cat44(*(const v4i*)(bp2 + bo), *(const v4i*)(bp2 + bo + 32));
                v8i bf3 = cat44(*(const v4i*)(bp3 + bo), *(const v4i*)(bp3 + bo + 32));
                acc[0][0] = __builtin_amdgcn_wmma_i32_16x16x64_iu8(true, af0, true, bf0, acc[0][0], false, false);
                acc[1][0] = __builtin_amdgcn_wmma_i32_16x16x64_iu8(true, af1, true, bf0, acc[1][0], false, false);
                acc[0][1] = __builtin_amdgcn_wmma_i32_16x16x64_iu8(true, af0, true, bf1, acc[0][1], false, false);
                acc[1][1] = __builtin_amdgcn_wmma_i32_16x16x64_iu8(true, af1, true, bf1, acc[1][1], false, false);
                acc[0][2] = __builtin_amdgcn_wmma_i32_16x16x64_iu8(true, af0, true, bf2, acc[0][2], false, false);
                acc[1][2] = __builtin_amdgcn_wmma_i32_16x16x64_iu8(true, af1, true, bf2, acc[1][2], false, false);
                acc[0][3] = __builtin_amdgcn_wmma_i32_16x16x64_iu8(true, af0, true, bf3, acc[0][3], false, false);
                acc[1][3] = __builtin_amdgcn_wmma_i32_16x16x64_iu8(true, af1, true, bf3, acc[1][3], false, false);
            }
            Ap0 += 3072; Ap1 += 3072;
            bp0 += 192; bp1 += 192; bp2 += 192; bp3 += 192;
        }
    }

#pragma unroll
    for (int mi = 0; mi < 2; mi++) {
        int obase = (tm2 * 2 + mi) * 16 + hi * 8;
        bool full = (obase + 8) <= M;
#pragma unroll
        for (int nt = 0; nt < 4; nt++) {
            if (!on[nt]) continue;
            int col = n0 + nt * 16 + lo;
            float* op = out + (long)col * M + obase;
            if (full) {
                const float4* bp4 = (const float4*)(bias + obase);
                float4 b0 = bp4[0], b1 = bp4[1];
                float4 v0, v1;
                v0.x = (float)acc[mi][nt][0] + b0.x;
                v0.y = (float)acc[mi][nt][1] + b0.y;
                v0.z = (float)acc[mi][nt][2] + b0.z;
                v0.w = (float)acc[mi][nt][3] + b0.w;
                v1.x = (float)acc[mi][nt][4] + b1.x;
                v1.y = (float)acc[mi][nt][5] + b1.y;
                v1.z = (float)acc[mi][nt][6] + b1.z;
                v1.w = (float)acc[mi][nt][7] + b1.w;
                *(float4*)op = v0;
                *(float4*)(op + 4) = v1;
            } else {
#pragma unroll
                for (int r = 0; r < 8; r++) {
                    int o = obase + r;
                    if (o < M) op[r] = (float)acc[mi][nt][r] + bias[o];
                }
            }
        }
    }
}

__global__ void __launch_bounds__(BDIM)
bn_logsoftmax_kernel(const float* __restrict__ h,
                     const float* __restrict__ mean, const float* __restrict__ invstd,
                     const float* __restrict__ g, const float* __restrict__ be,
                     float* __restrict__ out)
{
    __shared__ float red[BDIM];
    int b = blockIdx.x;
    int tid = threadIdx.x;
    const float* hb = h + (long)b * 1000;

    float mx = -3.4e38f;
    for (int c = tid; c < 1000; c += BDIM) {
        float y = (hb[c] - mean[c]) * invstd[c] * g[c] + be[c];
        mx = fmaxf(mx, y);
    }
    red[tid] = mx; __syncthreads();
    for (int st = BDIM / 2; st > 0; st >>= 1) {
        if (tid < st) red[tid] = fmaxf(red[tid], red[tid + st]);
        __syncthreads();
    }
    mx = red[0]; __syncthreads();

    float se = 0.f;
    for (int c = tid; c < 1000; c += BDIM) {
        float y = (hb[c] - mean[c]) * invstd[c] * g[c] + be[c];
        se += expf(y - mx);
    }
    red[tid] = se; __syncthreads();
    for (int st = BDIM / 2; st > 0; st >>= 1) {
        if (tid < st) red[tid] += red[tid + st];
        __syncthreads();
    }
    float lse = logf(red[0]);

    for (int c = tid; c < 1000; c += BDIM) {
        float y = (hb[c] - mean[c]) * invstd[c] * g[c] + be[c];
        out[(long)b * 1000 + c] = y - mx - lse;
    }
}

extern "C" void kernel_launch(void* const* d_in, const int* in_sizes, int n_in,
                              void* d_out, int out_size, void* d_ws, size_t ws_size,
                              hipStream_t stream)
{
    (void)in_sizes; (void)n_in; (void)out_size; (void)ws_size;
    const float* x   = (const float*)d_in[0];
    const float* w1  = (const float*)d_in[1];
    const float* w2  = (const float*)d_in[2];
    const float* w3  = (const float*)d_in[3];
    const float* w4  = (const float*)d_in[4];
    const float* w5  = (const float*)d_in[5];
    const float* fw1 = (const float*)d_in[6];
    const float* fw2 = (const float*)d_in[7];
    const float* b1  = (const float*)d_in[8];
    const float* b2  = (const float*)d_in[9];
    const float* b3  = (const float*)d_in[10];
    const float* b4  = (const float*)d_in[11];
    const float* b5  = (const float*)d_in[12];
    const float* fb1 = (const float*)d_in[13];
    const float* fb2 = (const float*)d_in[14];
    const float* be1 = (const float*)d_in[15];
    const float* be2 = (const float*)d_in[16];
    const float* be3 = (const float*)d_in[17];
    const float* be4 = (const float*)d_in[18];
    const float* be5 = (const float*)d_in[19];
    const float* be6 = (const float*)d_in[20];
    const float* be7 = (const float*)d_in[21];
    const float* g1  = (const float*)d_in[22];
    const float* g2  = (const float*)d_in[23];
    const float* g3  = (const float*)d_in[24];
    const float* g4  = (const float*)d_in[25];
    const float* g5  = (const float*)d_in[26];
    const float* g6  = (const float*)d_in[27];
    const float* g7  = (const float*)d_in[28];

    char* ws = (char*)d_ws;
    size_t off = 0;
    auto carve = [&](size_t bytes) -> char* {
        char* p = ws + off;
        off = (off + bytes + 255) & ~(size_t)255;
        return p;
    };
    float*       F0    = (float*)carve(117374976);
    float*       F1    = (float*)carve(51904512);
    signed char* S0    = (signed char*)carve(19501056);
    signed char* Wp    = (signed char*)carve(3317760);
    float*       meanb = (float*)carve(4608 * 4);
    float*       istdb = (float*)carve(4608 * 4);
    float*       fc1o  = (float*)carve(16 * 4608 * 4);
    float*       fc2o  = (float*)carve(16 * 1000 * 4);
    signed char* Wbig  = (signed char*)F0;

    dim3 blk(BDIM);
    auto gb = [](long n) { return dim3((unsigned)((n + BDIM - 1) / BDIM)); };
    auto packw = [&](const float* wsrc, signed char* dst, int O, int I, int T, int Opad) {
        long dw = ((long)Opad * I * T) >> 2;
        pack_w_swz_kernel<<<gb(dw), blk, 0, stream>>>(wsrc, dst, O, I, T, Opad);
    };
    auto gemm = [&](const signed char* A, const signed char* actp, const float* bias,
                    float* outp, int M, int Mpad, int Cin, int T, int dil,
                    int Lin, int Lout, int Ntot) {
        long wtiles = (long)(Mpad / 32) * ((Ntot + 63) / 64);
        bin_gemm_wmma<<<(unsigned)((wtiles + 7) / 8), blk, 0, stream>>>(
            A, actp, bias, outp, M, Mpad, Cin, T, dil, Lin, Lout, Ntot);
    };

    conv1_pool_kernel<<<gb(16L * 3196 * 192), blk, 0, stream>>>(x, w1, b1, F1);
    bn_stats_kernel<<<192, blk, 0, stream>>>(F1, 16L * 3196, 192, meanb, istdb);
    bn_relu_sign_kernel<<<gb(16L * 3196 * 192), blk, 0, stream>>>(
        F1, 16L * 3196 * 192, 192, meanb, istdb, g1, be1, S0);

    packw(w2, Wp, 576, 192, 5, 576);
    gemm(Wp, S0, b2, F0, 576, 576, 192, 5, 3, 3196, 3184, 16 * 3184);
    maxpool_kernel<<<gb(16L * 1062 * 576), blk, 0, stream>>>(F0, F1, 16, 3184, 1062, 576, 3, 1);
    bn_stats_kernel<<<576, blk, 0, stream>>>(F1, 16L * 1062, 576, meanb, istdb);
    bn_relu_sign_kernel<<<gb(16L * 1062 * 576), blk, 0, stream>>>(
        F1, 16L * 1062 * 576, 576, meanb, istdb, g2, be2, S0);

    packw(w3, Wp, 1152, 576, 5, 1152);
    gemm(Wp, S0, b3, F0, 1152, 1152, 576, 5, 1, 1062, 1058, 16 * 1058);
    bn_stats_kernel<<<1152, blk, 0, stream>>>(F0, 16L * 1058, 1152, meanb, istdb);
    bn_relu_sign_kernel<<<gb(16L * 1058 * 1152), blk, 0, stream>>>(
        F0, 16L * 1058 * 1152, 1152, meanb, istdb, g3, be3, S0);

    packw(w4, Wp, 768, 1152, 3, 768);
    gemm(Wp, S0, b4, F1, 768, 768, 1152, 3, 1, 1058, 1056, 16 * 1056);
    bn_stats_kernel<<<768, blk, 0, stream>>>(F1, 16L * 1056, 768, meanb, istdb);
    bn_relu_sign_kernel<<<gb(16L * 1056 * 768), blk, 0, stream>>>(
        F1, 16L * 1056 * 768, 768, meanb, istdb, g4, be4, S0);

    packw(w5, Wp, 72, 768, 3, 96);
    gemm(Wp, S0, b5, F0, 72, 96, 768, 3, 1, 1056, 1054, 16 * 1054);
    maxpool_kernel<<<gb(16L * 352 * 72), blk, 0, stream>>>(F0, F1, 16, 1054, 352, 72, 3, 1);
    bn_stats_kernel<<<72, blk, 0, stream>>>(F1, 16L * 352, 72, meanb, istdb);
    bn_relu_sign_tr5_kernel<<<gb(16L * 352 * 72), blk, 0, stream>>>(
        F1, meanb, istdb, g5, be5, S0);

    packw(fw1, Wbig, 4608, 25344, 1, 4608);
    gemm(Wbig, S0, fb1, fc1o, 4608, 4608, 25344, 1, 0, 1, 1, 16);
    bn_stats_kernel<<<4608, blk, 0, stream>>>(fc1o, 16, 4608, meanb, istdb);
    bn_relu_sign_kernel<<<gb(16L * 4608), blk, 0, stream>>>(
        fc1o, 16L * 4608, 4608, meanb, istdb, g6, be6, S0);

    packw(fw2, Wbig, 1000, 4608, 1, 1024);
    gemm(Wbig, S0, fb2, fc2o, 1000, 1024, 4608, 1, 0, 1, 1, 16);
    bn_stats_kernel<<<1000, blk, 0, stream>>>(fc2o, 16, 1000, meanb, istdb);
    bn_logsoftmax_kernel<<<16, blk, 0, stream>>>(fc2o, meanb, istdb, g7, be7, (float*)d_out);
}
